// GPT_15556371546950
// MI455X (gfx1250) — matrix operand order repaired and checked by reading, not compiled, not run on hardware
//
#include <hip/hip_runtime.h>
#include <hip/hip_bf16.h>
#include <math.h>

// ---------------- problem constants ----------------
#define T_    2048
#define C_    2048
#define HS_   128
#define NH_   16
#define NG_   4
#define QPK_  4
#define OQKV_ 3072           // (NH + 2*NG) * HS
#define NEG_  (-1000000000.0f)

typedef __bf16 bf16_t;
typedef __attribute__((ext_vector_type(16))) __bf16 bf16x16;
typedef __attribute__((ext_vector_type(8)))  __bf16 bf16x8;
typedef __attribute__((ext_vector_type(8)))  float  f32x8;
typedef __attribute__((ext_vector_type(4)))  float  f32x4;

__device__ __forceinline__ f32x8 wmma_bf16(bf16x16 a, bf16x16 b, f32x8 c) {
  return __builtin_amdgcn_wmma_f32_16x16x32_bf16(
      /*neg_a=*/false, a, /*neg_b=*/false, b,
      /*c_mod=*/(short)0, c, /*reuse_a=*/false, /*reuse_b=*/false);
}

// load 16 contiguous bf16 (two 16B loads; 16B aligned by construction)
__device__ __forceinline__ bf16x16 ld16(const bf16_t* p) {
  bf16x16 r;
  *(bf16x8*)&r       = *(const bf16x8*)p;
  *((bf16x8*)&r + 1) = *(const bf16x8*)(p + 8);
  return r;
}
// A-fragment gather: 16 contiguous halves at p (k = 16h + s)
__device__ __forceinline__ bf16x16 ld8x2(const bf16_t* p) {
  bf16x16 r;
  *(bf16x8*)&r       = *(const bf16x8*)p;
  *((bf16x8*)&r + 1) = *(const bf16x8*)(p + 8);
  return r;
}

// CDNA5 async global->LDS copy (GVS mode), tracked with ASYNCcnt.
__device__ __forceinline__ void async_load_b128(unsigned lds_off, unsigned goff,
                                                const void* base) {
  asm volatile("global_load_async_to_lds_b128 %0, %1, %2"
               :: "v"(lds_off), "v"(goff), "s"(base) : "memory");
}
__device__ __forceinline__ void wait_async0() {
  asm volatile("s_wait_asynccnt 0x0" ::: "memory");
}
// allow one tile (8 copies/wave) in flight: async loads retire in order,
// so <=8 outstanding means the *previous* tile is fully resident.
__device__ __forceinline__ void wait_async8() {
  asm volatile("s_wait_asynccnt 0x8" ::: "memory");
}

// =====================================================================
// fp32 -> bf16 one-shot convert (read once, halve all downstream traffic)
// =====================================================================
__global__ __launch_bounds__(256) void cvt_f32_bf16(const float* __restrict__ in,
                                                    bf16_t* __restrict__ out) {
  const size_t i = ((size_t)blockIdx.x * 256 + threadIdx.x) * 4;
  const f32x4 v = *(const f32x4*)(in + i);
  bf16_t* o = out + i;
  o[0] = (bf16_t)v.x; o[1] = (bf16_t)v.y; o[2] = (bf16_t)v.z; o[3] = (bf16_t)v.w;
}

// =====================================================================
// GEMM: out[m][n] = sum_k A[m][k] * W[n][k] + bias[n]  (opt ReLU)
// A: M x K bf16 row-major, W: N x K bf16 row-major. 128x128x64 tiles,
// 256 threads = 8 wave32; each wave computes a 64x32 sub-tile (4x2 frags).
// Double-buffered LDS; tiles staged with GLOBAL_LOAD_ASYNC_TO_LDS_B128 so
// HBM/L2 latency of tile t+1 hides behind the 16 WMMAs of tile t.
// =====================================================================
__global__ __launch_bounds__(256) void gemm_wmma_bf16(
    const bf16_t* __restrict__ A, const bf16_t* __restrict__ W,
    const float* __restrict__ bias, float* __restrict__ out,
    int N, int K, int relu)
{
  constexpr int STRIDE = 72;            // halves: 64 + 8 pad -> 144B (16B-mult)
  constexpr unsigned TILE_B = 128u * STRIDE * 2u;  // bytes per buffer
  __shared__ bf16_t As[2][128 * STRIDE];
  __shared__ bf16_t Ws[2][128 * STRIDE];

  const int tid  = threadIdx.x;
  const int lane = tid & 31;
  const int wid  = tid >> 5;
  const int wm   = wid & 1;             // 2 waves along M (64 rows each)
  const int wn   = wid >> 1;            // 4 waves along N (32 cols each)
  const int bm0  = blockIdx.x * 128;
  const int bn0  = blockIdx.y * 128;
  const int l16  = lane & 15;
  const int half8  = (lane < 16) ? 0 : 8;
  const int half16 = (lane < 16) ? 0 : 16;

  f32x8 acc[4][2] = {};

  // per-thread copy plan: 4 b128 chunks per matrix per tile
  // chunk c = tid + i*256: row = c>>3, colchunk = c&7 (8 chunks per 64-col row)
  const unsigned a_base = (unsigned)(uintptr_t)&As[0][0];
  const unsigned w_base = (unsigned)(uintptr_t)&Ws[0][0];
  unsigned a_loff[4], w_loff[4], a_goff[4], w_goff[4];
  #pragma unroll
  for (int i = 0; i < 4; ++i) {
    const int c = tid + i * 256;
    const int row = c >> 3, cc = c & 7;
    a_loff[i] = a_base + row * (STRIDE * 2) + cc * 16;
    w_loff[i] = w_base + row * (STRIDE * 2) + cc * 16;
    a_goff[i] = (unsigned)(bm0 + row) * (unsigned)(K * 2) + cc * 16;
    w_goff[i] = (unsigned)(bn0 + row) * (unsigned)(K * 2) + cc * 16;
  }

  const int NT = K >> 6;                // K / 64 tiles

  // prologue: tile 0 -> buffer 0
  #pragma unroll
  for (int i = 0; i < 4; ++i) {
    async_load_b128(a_loff[i], a_goff[i], A);
    async_load_b128(w_loff[i], w_goff[i], W);
  }

  for (int t = 0; t < NT; ++t) {
    const int cur = t & 1;
    if (t + 1 < NT) {                   // prefetch tile t+1 into other buffer
      const unsigned kb = (unsigned)((t + 1) << 7);   // *64 cols *2 bytes
      const unsigned bo = (cur ^ 1) ? TILE_B : 0u;
      #pragma unroll
      for (int i = 0; i < 4; ++i) {
        async_load_b128(a_loff[i] + bo, a_goff[i] + kb, A);
        async_load_b128(w_loff[i] + bo, w_goff[i] + kb, W);
      }
      wait_async8();                    // tile t resident (in-order retire)
    } else {
      wait_async0();
    }
    __syncthreads();                    // tile t visible to all waves

    const bf16_t* at = As[cur];
    const bf16_t* wt = Ws[cur];
    #pragma unroll
    for (int ks = 0; ks < 2; ++ks) {    // two K=32 steps per tile
      bf16x16 bfrag[2];
      #pragma unroll
      for (int fn = 0; fn < 2; ++fn)    // B: lane=col n, K 0..15 / 16..31
        bfrag[fn] = ld16(wt + (wn * 32 + fn * 16 + l16) * STRIDE
                            + ks * 32 + half16);
      #pragma unroll
      for (int fm = 0; fm < 4; ++fm) {  // A: lane=row m, K 0..15 / 16..31
        bf16x16 afrag = ld8x2(at + (wm * 64 + fm * 16 + l16) * STRIDE
                                 + ks * 32 + half16);
        #pragma unroll
        for (int fn = 0; fn < 2; ++fn)
          acc[fm][fn] = wmma_bf16(afrag, bfrag[fn], acc[fm][fn]);
      }
    }
    __syncthreads();                    // seal reads before buffer reuse
  }

  const int mhalf = (lane < 16) ? 0 : 8;
  #pragma unroll
  for (int fm = 0; fm < 4; ++fm)
    #pragma unroll
    for (int fn = 0; fn < 2; ++fn) {
      const int col = bn0 + wn * 32 + fn * 16 + l16;
      const float bv = bias[col];
      #pragma unroll
      for (int j = 0; j < 8; ++j) {
        const int row = bm0 + wm * 64 + fm * 16 + mhalf + j;
        float v = acc[fm][fn][j] + bv;
        if (relu) v = v > 0.f ? v : 0.f;
        out[(size_t)row * N + col] = v;
      }
    }
}

// =====================================================================
// RoPE + repack: qkv f32 (T x 3072) ->
//   q bf16 (NH,T,HS) roped; k bf16 (NG,T,HS) roped; vT bf16 (NG,HS,T)
// =====================================================================
__global__ void rope_pack(const float* __restrict__ qkv,
                          const float* __restrict__ cosb,
                          const float* __restrict__ sinb,
                          bf16_t* __restrict__ qd,
                          bf16_t* __restrict__ kd,
                          bf16_t* __restrict__ vt)
{
  const int idx = blockIdx.x * blockDim.x + threadIdx.x;
  if (idx >= T_ * OQKV_) return;
  const int t = idx / OQKV_;
  const int o = idx % OQKV_;
  const int g = o / ((QPK_ + 2) * HS_);
  const int r = o % ((QPK_ + 2) * HS_);
  const int j = r / HS_;
  const int d = r % HS_;
  const float v = qkv[idx];
  if (j <= QPK_) {                      // q heads + k: apply RoPE
    const int   pd  = (d < HS_ / 2) ? d + HS_ / 2 : d - HS_ / 2;
    const float sgn = (d < HS_ / 2) ? -1.f : 1.f;
    const float rot = sgn * qkv[(size_t)t * OQKV_ + (o - d) + pd];
    const float rv  = v * cosb[t * HS_ + d] + rot * sinb[t * HS_ + d];
    if (j < QPK_) {
      const int h = g * QPK_ + j;
      qd[((size_t)h * T_ + t) * HS_ + d] = (bf16_t)rv;
    } else {
      kd[((size_t)g * T_ + t) * HS_ + d] = (bf16_t)rv;
    }
  } else {                              // v: store transposed (g, d, t)
    vt[((size_t)g * HS_ + d) * T_ + t] = (bf16_t)v;
  }
}

// =====================================================================
// Wave-level flash attention. Each wave32 owns 16 query rows of one head.
// Block = 8 waves = 128 query rows. Keys streamed in chunks of 32.
// logits = (q.k * sf + causal_bias) * scaling[q,k]; online softmax.
// Output written directly as bf16 for the projection GEMM.
// =====================================================================
__global__ __launch_bounds__(256) void attn_wmma(
    const bf16_t* __restrict__ q, const bf16_t* __restrict__ k,
    const bf16_t* __restrict__ vt, const float* __restrict__ scal,
    bf16_t* __restrict__ y)
{
  __shared__ bf16_t Pst[8][16 * 40];    // per-wave P staging (16 x 32, pad 40)
  const int lane  = threadIdx.x & 31;
  const int w     = threadIdx.x >> 5;
  const int h     = blockIdx.y;
  const int g     = h >> 2;             // h / QPK
  const int q0    = blockIdx.x * 128 + w * 16;
  const int l16   = lane & 15;
  const int khalf = (lane < 16) ? 0 : 1;
  const int mhalf = (lane < 16) ? 0 : 8;
  const float sf  = 0.08838834764831845f;  // 1/sqrt(128)

  // Q A-fragments (16 rows x 128 feat = 4 k-steps of 32), direct global loads
  bf16x16 qa[4];
  {
    const bf16_t* qrow = q + ((size_t)h * T_ + q0 + l16) * HS_;
    #pragma unroll
    for (int ks = 0; ks < 4; ++ks)
      qa[ks] = ld8x2(qrow + ks * 32 + khalf * 16);
  }

  f32x8 o[8] = {};
  float mrow[8], lrow[8];
  #pragma unroll
  for (int j = 0; j < 8; ++j) { mrow[j] = -3.0e38f; lrow[j] = 0.f; }

  const int kend = q0 + 16;             // causal: need keys < q0+16
  for (int k0 = 0; k0 < kend; k0 += 32) {
    // S = Q K^T for 32 keys (two 16-wide N chunks)
    f32x8 s[2];
    #pragma unroll
    for (int kk = 0; kk < 2; ++kk) {
      f32x8 sacc = {};
      #pragma unroll
      for (int ks = 0; ks < 4; ++ks) {
        const bf16_t* kp = k + ((size_t)g * T_ + k0 + kk * 16 + l16) * HS_
                             + ks * 32 + khalf * 16;
        sacc = wmma_bf16(qa[ks], ld16(kp), sacc);
      }
      s[kk] = sacc;
    }

    // logits: scale, causal bias, per-(q,k) gate
    #pragma unroll
    for (int kk = 0; kk < 2; ++kk) {
      const int kcol = k0 + kk * 16 + l16;
      #pragma unroll
      for (int j = 0; j < 8; ++j) {
        const int qq = q0 + mhalf + j;
        const float bias = (kcol <= qq) ? 0.f : NEG_;
        const float sc   = scal[(size_t)qq * T_ + kcol];
        s[kk][j] = (s[kk][j] * sf + bias) * sc;
      }
    }

    // online softmax (row = VGPR index within 16-lane half)
    float corr[8];
    #pragma unroll
    for (int j = 0; j < 8; ++j) {
      float r = fmaxf(s[0][j], s[1][j]);
      r = fmaxf(r, __shfl_xor(r, 1));
      r = fmaxf(r, __shfl_xor(r, 2));
      r = fmaxf(r, __shfl_xor(r, 4));
      r = fmaxf(r, __shfl_xor(r, 8));
      const float nm = fmaxf(mrow[j], r);
      const float p0 = __expf(s[0][j] - nm);
      const float p1 = __expf(s[1][j] - nm);
      float rs = p0 + p1;
      rs += __shfl_xor(rs, 1);
      rs += __shfl_xor(rs, 2);
      rs += __shfl_xor(rs, 4);
      rs += __shfl_xor(rs, 8);
      corr[j]  = __expf(mrow[j] - nm);
      lrow[j]  = lrow[j] * corr[j] + rs;
      mrow[j]  = nm;
      const int m = mhalf + j;
      Pst[w][m * 40 + l16]      = (bf16_t)p0;   // D-layout -> row-major LDS
      Pst[w][m * 40 + 16 + l16] = (bf16_t)p1;
    }
    #pragma unroll
    for (int fn = 0; fn < 8; ++fn)
      #pragma unroll
      for (int j = 0; j < 8; ++j) o[fn][j] *= corr[j];

    // P A-fragment back from LDS (same-wave DS dependency; no barrier)
    bf16x16 pa = ld8x2(&Pst[w][l16 * 40 + khalf * 16]);

    // O += P x V  (V transposed: B-frag lane = feature col, keys contiguous)
    #pragma unroll
    for (int fn = 0; fn < 8; ++fn) {
      const bf16_t* vp = vt + ((size_t)g * HS_ + fn * 16 + l16) * T_
                            + k0 + khalf * 16;
      o[fn] = wmma_bf16(pa, ld16(vp), o[fn]);
    }
  }

  // normalize + write y[t][h*HS + d] as bf16 (feeds projection GEMM)
  #pragma unroll
  for (int fn = 0; fn < 8; ++fn)
    #pragma unroll
    for (int j = 0; j < 8; ++j) {
      const int qq = q0 + mhalf + j;
      const int cc = h * HS_ + fn * 16 + l16;
      y[(size_t)qq * C_ + cc] = (bf16_t)(o[fn][j] / lrow[j]);
    }
}

// =====================================================================
extern "C" void kernel_launch(void* const* d_in, const int* in_sizes, int n_in,
                              void* d_out, int out_size, void* d_ws, size_t ws_size,
                              hipStream_t stream) {
  (void)in_sizes; (void)n_in; (void)out_size; (void)ws_size;
  const float* x       = (const float*)d_in[0];
  const float* cosb    = (const float*)d_in[1];
  const float* sinb    = (const float*)d_in[2];
  const float* w_qkv   = (const float*)d_in[3];
  const float* b_qkv   = (const float*)d_in[4];
  const float* w_proj  = (const float*)d_in[5];
  const float* b_proj  = (const float*)d_in[6];
  const float* w_scale = (const float*)d_in[7];
  const float* b_scale = (const float*)d_in[8];
  float* out = (float*)d_out;

  char* ws = (char*)d_ws;
  float*  qkv  = (float*)ws;  ws += (size_t)T_ * OQKV_ * sizeof(float);
  float*  scal = (float*)ws;  ws += (size_t)T_ * T_ * sizeof(float);
  bf16_t* xb   = (bf16_t*)ws; ws += (size_t)T_ * C_ * sizeof(bf16_t);
  bf16_t* wqb  = (bf16_t*)ws; ws += (size_t)OQKV_ * C_ * sizeof(bf16_t);
  bf16_t* wsb  = (bf16_t*)ws; ws += (size_t)C_ * C_ * sizeof(bf16_t);
  bf16_t* wpb  = (bf16_t*)ws; ws += (size_t)C_ * C_ * sizeof(bf16_t);
  bf16_t* yb   = (bf16_t*)ws; ws += (size_t)T_ * C_ * sizeof(bf16_t);
  bf16_t* qd   = (bf16_t*)ws; ws += (size_t)NH_ * T_ * HS_ * sizeof(bf16_t);
  bf16_t* kd   = (bf16_t*)ws; ws += (size_t)NG_ * T_ * HS_ * sizeof(bf16_t);
  bf16_t* vt   = (bf16_t*)ws; ws += (size_t)NG_ * T_ * HS_ * sizeof(bf16_t);

  // 0) one-shot fp32 -> bf16 converts (read-once; halves GEMM traffic)
  cvt_f32_bf16<<<(T_ * C_) / 1024,     256, 0, stream>>>(x,       xb);
  cvt_f32_bf16<<<(OQKV_ * C_) / 1024,  256, 0, stream>>>(w_qkv,   wqb);
  cvt_f32_bf16<<<(C_ * C_) / 1024,     256, 0, stream>>>(w_scale, wsb);
  cvt_f32_bf16<<<(C_ * C_) / 1024,     256, 0, stream>>>(w_proj,  wpb);

  // 1) qkv = x @ w_qkv^T + b_qkv           (2048 x 3072)
  gemm_wmma_bf16<<<dim3(T_ / 128, OQKV_ / 128), 256, 0, stream>>>(
      xb, wqb, b_qkv, qkv, OQKV_, C_, 0);
  // 2) scaling = relu(x @ w_scale^T + b)   (2048 x 2048)
  gemm_wmma_bf16<<<dim3(T_ / 128, T_ / 128), 256, 0, stream>>>(
      xb, wsb, b_scale, scal, T_, C_, 1);
  // 3) RoPE + bf16 repack (q, k roped; v transposed)
  rope_pack<<<(T_ * OQKV_ + 255) / 256, 256, 0, stream>>>(
      qkv, cosb, sinb, qd, kd, vt);
  // 4) gated causal flash attention -> y bf16 (2048 x 2048)
  attn_wmma<<<dim3(T_ / 128, NH_), 256, 0, stream>>>(qd, kd, vt, scal, yb);
  // 5) out = y @ w_proj^T + b_proj
  gemm_wmma_bf16<<<dim3(T_ / 128, C_ / 128), 256, 0, stream>>>(
      yb, wpb, b_proj, out, C_, C_, 0);
}
